// MaskHDR_65249143161337
// MI455X (gfx1250) — matrix operand order repaired and checked by reading, not compiled, not run on hardware
//
#include <hip/hip_runtime.h>

// MaskHDR fused pipeline for MI455X (gfx1250, wave32).
// Memory-bound (~1.9 GB traffic @ 23.3 TB/s); blur done via v_wmma_f32_16x16x32_f16
// with banded 0.2 matrices (separable 5x5 avg), morphology via byte-mask AND/OR.

typedef __attribute__((ext_vector_type(16))) _Float16 v16h;
typedef __attribute__((ext_vector_type(8)))  _Float16 v8h;
typedef __attribute__((ext_vector_type(8)))  float    v8f;

namespace {
constexpr int Bn = 16, Cn = 3, Hn = 1024, Wn = 1024;
constexpr int PLANE = Hn * Wn;                 // 2^20
constexpr int TX = 4, TY = 2;                  // 16x16 tiles per block -> 64x32 region
constexpr int SW = 16 * TX + 16;               // 80 strip cols (f16)
constexpr int SH = 16 * TY + 16;               // 48 strip rows
constexpr int VALID_W = 16 * TX + 4;           // 68 real cols (halo 2)
constexpr int VALID_H = 16 * TY + 4;           // 36 real rows
constexpr float THR = 0.3f;
// stats layout (float index): [6]=factor [7]=min_nr [8]=max_nr [9]=min_m [10]=max_m
constexpr int OFF_PSUM_NR = 16, OFF_PSUM_R = 272, OFF_PMIN_NR = 528,
              OFF_PMAX_NR = 784, OFF_PMIN_M = 1040, OFF_PMAX_M = 1296;
}

// ---------------- wave reductions (wave32) ----------------
__device__ inline float wredSum(float v) {
  #pragma unroll
  for (int o = 16; o > 0; o >>= 1) v += __shfl_down(v, o);
  return v;
}
__device__ inline float wredMin(float v) {
  #pragma unroll
  for (int o = 16; o > 0; o >>= 1) v = fminf(v, __shfl_down(v, o));
  return v;
}
__device__ inline float wredMax(float v) {
  #pragma unroll
  for (int o = 16; o > 0; o >>= 1) v = fmaxf(v, __shfl_down(v, o));
  return v;
}

// ---------------- WMMA blur pieces ----------------
// B-matrix (32x16 f16) layout: lane n=L&15, lh=L>>4; e<8 -> K=e+8*lh, e>=8 -> K=e+8+8*lh.
// Horizontal band: B[k][n] = 0.2 for k in [n, n+4].
__device__ inline v16h make_Bh(int lane) {
  const int n = lane & 15, lh = lane >> 4;
  v16h b;
  #pragma unroll
  for (int e = 0; e < 16; ++e) {
    int k = e + 8 * lh + (e >= 8 ? 8 : 0);
    b[e] = (_Float16)((k >= n && k <= n + 4) ? 0.2f : 0.0f);
  }
  return b;
}
// A2 (16x32 f16) uses contiguous k, as the b2 reload does: lane M=L&15, lh=L>>4; element e -> K = e + 16*lh.
// Vertical band: A[m][k] = 0.2 for k in [m, m+4].
__device__ inline v16h make_A2(int lane) {
  const int m = lane & 15, lh = lane >> 4;
  v16h a;
  #pragma unroll
  for (int e = 0; e < 16; ++e) {
    int k = e + 16 * lh;
    a[e] = (_Float16)((k >= m && k <= m + 4) ? 0.2f : 0.0f);
  }
  return a;
}

__device__ inline v16h load_A16x32(const _Float16* __restrict__ strip, int r0, int c0, int lane) {
  const int M = lane & 15, lh = lane >> 4;
  const _Float16* p = strip + (r0 + M) * SW + c0 + 8 * lh;  // 16B aligned
  union { v16h v; v8h h[2]; } u;
  u.h[0] = *(const v8h*)p;          // elements 0..7  : K = 8*lh + 0..7
  u.h[1] = *(const v8h*)(p + 16);   // elements 8..15 : K = 16 + 8*lh + 0..7
  return u.v;
}

// Full separable 5x5 avg blur of one 16x16 tile. tb = per-wave 512-half LDS scratch.
// Returns D layout: VGPR r holds pixel (M = r + 8*(lane>>4), N = lane&15).
__device__ inline v8f blur_tile(const _Float16* __restrict__ strip, _Float16* __restrict__ tb,
                                int r0, int c0, int lane, v16h Bh, v16h A2) {
  v8f z = {};
  const int n = lane & 15, lh = lane >> 4;
  #pragma unroll
  for (int h = 0; h < 2; ++h) {
    v16h A = load_A16x32(strip, r0 + 16 * h, c0, lane);
    v8f T = __builtin_amdgcn_wmma_f32_16x16x32_f16(false, A, false, Bh, (short)0, z, false, false);
    // store transposed: T_t[N][16*h + 8*lh + r], 8 contiguous halves -> one 16B ds store
    v8h pk;
    #pragma unroll
    for (int r = 0; r < 8; ++r) pk[r] = (_Float16)T[r];
    *(v8h*)(tb + n * 32 + 16 * h + 8 * lh) = pk;
  }
  // reload as B-matrix: lane n, K = 16*lh + 0..15 contiguous in T_t row n
  union { v16h v; v8h h[2]; } b2;
  const _Float16* q = tb + n * 32 + 16 * lh;
  b2.h[0] = *(const v8h*)q;
  b2.h[1] = *(const v8h*)(q + 8);
  return __builtin_amdgcn_wmma_f32_16x16x32_f16(false, A2, false, b2.v, (short)0, z, false, false);
}

// Stage (gy0-2..+SH-3) x (gx0-2..+SW-3) into f16 strip; zero outside image and
// beyond the real halo (those K rows/cols hit zero band entries -> must be finite).
__device__ inline void stage_strip(_Float16* __restrict__ strip, const float* __restrict__ img,
                                   int gy0, int gx0) {
  for (int i = threadIdx.x; i < SH * SW; i += 256) {
    int s = i / SW, t = i - s * SW;
    int gy = gy0 - 2 + s, gx = gx0 - 2 + t;
    float v = 0.0f;
    if (s < VALID_H && t < VALID_W && gy >= 0 && gy < Hn && gx >= 0 && gx < Wn)
      v = img[(size_t)gy * Wn + gx];
    strip[i] = (_Float16)v;
  }
}

// ---------------- kernels ----------------
__global__ __launch_bounds__(256) void hdr_init(float* stats) {
  unsigned* u = (unsigned*)stats;
  int t = threadIdx.x;
  if (t < 16) stats[t] = 0.0f;
  stats[OFF_PSUM_NR + t] = 0.0f;
  stats[OFF_PSUM_R + t]  = 0.0f;
  u[OFF_PMIN_NR + t] = 0x7F800000u;  // +inf
  u[OFF_PMAX_NR + t] = 0u;
  u[OFF_PMIN_M + t]  = 0x7F800000u;
  u[OFF_PMAX_M + t]  = 0u;
}

__global__ __launch_bounds__(256) void hdr_stats(const float* __restrict__ nr,
                                                 const float* __restrict__ rf,
                                                 float* __restrict__ stats) {
  __shared__ __align__(16) _Float16 sN[SH * SW];
  __shared__ __align__(16) _Float16 sR[SH * SW];
  __shared__ __align__(16) _Float16 tb[8 * 512];
  const int plane = blockIdx.z;
  const int gx0 = blockIdx.x * (16 * TX), gy0 = blockIdx.y * (16 * TY);
  stage_strip(sN, nr + (size_t)plane * PLANE, gy0, gx0);
  stage_strip(sR, rf + (size_t)plane * PLANE, gy0, gx0);
  __syncthreads();
  const int lane = threadIdx.x & 31, wid = threadIdx.x >> 5;
  const int tx = wid & 3, ty = wid >> 2;
  v16h Bh = make_Bh(lane), A2 = make_A2(lane);
  _Float16* tw = tb + wid * 512;
  v8f bn = blur_tile(sN, tw, 16 * ty, 16 * tx, lane, Bh, A2);
  v8f br = blur_tile(sR, tw, 16 * ty, 16 * tx, lane, Bh, A2);
  float s1 = 0.f, s2 = 0.f, mn = 1e30f, mx = -1e30f;
  #pragma unroll
  for (int r = 0; r < 8; ++r) {
    s1 += bn[r]; s2 += br[r];
    mn = fminf(mn, bn[r]); mx = fmaxf(mx, bn[r]);
  }
  s1 = wredSum(s1); s2 = wredSum(s2); mn = wredMin(mn); mx = wredMax(mx);
  if (lane == 0) {
    int flat = (blockIdx.z * gridDim.y + blockIdx.y) * gridDim.x + blockIdx.x;
    int cell = (flat * 8 + wid) & 255;
    unsigned* u = (unsigned*)stats;
    atomicAdd(&stats[OFF_PSUM_NR + cell], s1);
    atomicAdd(&stats[OFF_PSUM_R + cell], s2);
    atomicMin(&u[OFF_PMIN_NR + cell], __float_as_uint(mn));  // values >= 0
    atomicMax(&u[OFF_PMAX_NR + cell], __float_as_uint(mx));
  }
}

__global__ __launch_bounds__(256) void hdr_reduce1(float* stats) {
  unsigned* u = (unsigned*)stats;
  const int t = threadIdx.x, lane = t & 31, wid = t >> 5;
  float s1 = stats[OFF_PSUM_NR + t], s2 = stats[OFF_PSUM_R + t];
  float mn = __uint_as_float(u[OFF_PMIN_NR + t]);
  float mx = __uint_as_float(u[OFF_PMAX_NR + t]);
  __shared__ float a1[8], a2[8], a3[8], a4[8];
  s1 = wredSum(s1); s2 = wredSum(s2); mn = wredMin(mn); mx = wredMax(mx);
  if (lane == 0) { a1[wid] = s1; a2[wid] = s2; a3[wid] = mn; a4[wid] = mx; }
  __syncthreads();
  if (t == 0) {
    float S1 = 0.f, S2 = 0.f, MN = 1e30f, MX = -1e30f;
    for (int i = 0; i < 8; ++i) {
      S1 += a1[i]; S2 += a2[i]; MN = fminf(MN, a3[i]); MX = fmaxf(MX, a4[i]);
    }
    stats[6] = S2 / S1;  // factor = mean(r)/mean(nr)
    stats[7] = MN;
    stats[8] = MX;
  }
}

__global__ __launch_bounds__(256) void hdr_mstat(const float* __restrict__ nr,
                                                 float* __restrict__ stats) {
  __shared__ __align__(16) _Float16 sN[SH * SW];
  __shared__ __align__(16) _Float16 tb[8 * 512];
  const int plane = blockIdx.z;
  const int gx0 = blockIdx.x * (16 * TX), gy0 = blockIdx.y * (16 * TY);
  stage_strip(sN, nr + (size_t)plane * PLANE, gy0, gx0);
  __syncthreads();
  const int lane = threadIdx.x & 31, wid = threadIdx.x >> 5;
  const int tx = wid & 3, ty = wid >> 2;
  v16h Bh = make_Bh(lane), A2 = make_A2(lane);
  v8f bn = blur_tile(sN, tb + wid * 512, 16 * ty, 16 * tx, lane, Bh, A2);
  const float factor = stats[6];
  float mn = 1e30f, mx = -1e30f;
  #pragma unroll
  for (int r = 0; r < 8; ++r) {
    float m0 = fminf(fmaxf(bn[r] * factor, 0.0f), 1.0f);
    mn = fminf(mn, m0); mx = fmaxf(mx, m0);
  }
  mn = wredMin(mn); mx = wredMax(mx);
  if (lane == 0) {
    int flat = (blockIdx.z * gridDim.y + blockIdx.y) * gridDim.x + blockIdx.x;
    int cell = (flat * 8 + wid) & 255;
    unsigned* u = (unsigned*)stats;
    atomicMin(&u[OFF_PMIN_M + cell], __float_as_uint(mn));
    atomicMax(&u[OFF_PMAX_M + cell], __float_as_uint(mx));
  }
}

__global__ __launch_bounds__(256) void hdr_reduce2(float* stats) {
  unsigned* u = (unsigned*)stats;
  const int t = threadIdx.x, lane = t & 31, wid = t >> 5;
  float mn = __uint_as_float(u[OFF_PMIN_M + t]);
  float mx = __uint_as_float(u[OFF_PMAX_M + t]);
  __shared__ float a3[8], a4[8];
  mn = wredMin(mn); mx = wredMax(mx);
  if (lane == 0) { a3[wid] = mn; a4[wid] = mx; }
  __syncthreads();
  if (t == 0) {
    float MN = 1e30f, MX = -1e30f;
    for (int i = 0; i < 8; ++i) { MN = fminf(MN, a3[i]); MX = fmaxf(MX, a4[i]); }
    stats[9] = MN;
    stats[10] = MX;
  }
}

__global__ __launch_bounds__(256) void hdr_mask(const float* __restrict__ nr,
                                                const float* __restrict__ rf,
                                                const float* __restrict__ stats,
                                                unsigned char* __restrict__ mask) {
  __shared__ __align__(16) _Float16 sN[SH * SW];
  __shared__ __align__(16) _Float16 sR[SH * SW];
  __shared__ __align__(16) _Float16 tb[8 * 512];
  const int b = blockIdx.z;
  const int gx0 = blockIdx.x * (16 * TX), gy0 = blockIdx.y * (16 * TY);
  const int lane = threadIdx.x & 31, wid = threadIdx.x >> 5;
  const int tx = wid & 3, ty = wid >> 2;
  const float factor = stats[6], mnr = stats[7], mxr = stats[8];
  const float mnm = stats[9], mxm = stats[10];
  const float scale = (mxr - mnr) / (mxm - mnm);
  v16h Bh = make_Bh(lane), A2 = make_A2(lane);
  _Float16* tw = tb + wid * 512;
  int hit[8] = {0, 0, 0, 0, 0, 0, 0, 0};
  for (int c = 0; c < Cn; ++c) {
    __syncthreads();  // strips are re-staged each channel
    stage_strip(sN, nr + (size_t)(b * Cn + c) * PLANE, gy0, gx0);
    stage_strip(sR, rf + (size_t)(b * Cn + c) * PLANE, gy0, gx0);
    __syncthreads();
    v8f bn = blur_tile(sN, tw, 16 * ty, 16 * tx, lane, Bh, A2);
    v8f br = blur_tile(sR, tw, 16 * ty, 16 * tx, lane, Bh, A2);
    #pragma unroll
    for (int r = 0; r < 8; ++r) {
      float m0 = fminf(fmaxf(bn[r] * factor, 0.0f), 1.0f);
      float nrm = (m0 - mnm) * scale + mnr;
      hit[r] |= (fabsf(nrm - br[r]) > THR) ? 1 : 0;
    }
  }
  const int n = lane & 15, lh = lane >> 4;
  const size_t bbase = (size_t)b * PLANE;
  #pragma unroll
  for (int r = 0; r < 8; ++r) {
    int gy = gy0 + 16 * ty + 8 * lh + r;   // D layout: M = r + 8*lh
    int gx = gx0 + 16 * tx + n;
    mask[bbase + (size_t)gy * Wn + gx] = (unsigned char)hit[r];
  }
}

// Separable 11x11 morphology on binary byte mask. OP=0 erode(AND, border ignored
// == cv2 +inf border), OP=1 dilate(OR, border ignored == cv2 -inf border).
template <int OP>
__global__ __launch_bounds__(256) void hdr_morph_h(const unsigned char* __restrict__ in,
                                                   unsigned char* __restrict__ out) {
  int idx = blockIdx.x * 256 + threadIdx.x;
  int x = idx & (Wn - 1);
  int base = idx - x;
  int lo = x - 5 < 0 ? 0 : x - 5;
  int hi = x + 5 > Wn - 1 ? Wn - 1 : x + 5;
  unsigned v = OP ? 0u : 1u;
  for (int xx = lo; xx <= hi; ++xx) {
    unsigned m = in[base + xx];
    v = OP ? (v | m) : (v & m);
  }
  out[idx] = (unsigned char)v;
}

template <int OP>
__global__ __launch_bounds__(256) void hdr_morph_v(const unsigned char* __restrict__ in,
                                                   unsigned char* __restrict__ out) {
  int idx = blockIdx.x * 256 + threadIdx.x;
  int y = (idx >> 10) & (Hn - 1);
  int colbase = idx - y * Wn;
  int lo = y - 5 < 0 ? 0 : y - 5;
  int hi = y + 5 > Hn - 1 ? Hn - 1 : y + 5;
  unsigned v = OP ? 0u : 1u;
  for (int yy = lo; yy <= hi; ++yy) {
    unsigned m = in[colbase + yy * Wn];
    v = OP ? (v | m) : (v & m);
  }
  out[idx] = (unsigned char)v;
}

// Fused vertical dilate + channel broadcast + (ghost, 1-ghost) writeout.
__global__ __launch_bounds__(256) void hdr_final(const unsigned char* __restrict__ m,
                                                 float* __restrict__ out) {
  int idx = blockIdx.x * 256 + threadIdx.x;
  int x = idx & (Wn - 1);
  int y = (idx >> 10) & (Hn - 1);
  int b = idx >> 20;
  int colbase = idx - y * Wn;
  int lo = y - 5 < 0 ? 0 : y - 5;
  int hi = y + 5 > Hn - 1 ? Hn - 1 : y + 5;
  unsigned v = 0u;
  for (int yy = lo; yy <= hi; ++yy) v |= m[colbase + yy * Wn];
  float g = v ? 1.0f : 0.0f;
  const size_t plane = (size_t)PLANE;
  const size_t Nout = (size_t)Bn * Cn * plane;
  size_t base = (size_t)b * Cn * plane + (size_t)y * Wn + x;
  out[base] = g;
  out[base + plane] = g;
  out[base + 2 * plane] = g;
  out[Nout + base] = 1.0f - g;
  out[Nout + base + plane] = 1.0f - g;
  out[Nout + base + 2 * plane] = 1.0f - g;
}

extern "C" void kernel_launch(void* const* d_in, const int* in_sizes, int n_in,
                              void* d_out, int out_size, void* d_ws, size_t ws_size,
                              hipStream_t stream) {
  (void)in_sizes; (void)n_in; (void)out_size; (void)ws_size;
  const float* nr = (const float*)d_in[0];
  const float* rf = (const float*)d_in[1];
  float* out = (float*)d_out;
  float* stats = (float*)d_ws;
  unsigned char* mA = (unsigned char*)d_ws + 65536;
  unsigned char* mB = mA + (size_t)Bn * PLANE;

  hdr_init<<<1, 256, 0, stream>>>(stats);

  dim3 gs(Wn / (16 * TX), Hn / (16 * TY), Bn * Cn);  // 16 x 32 x 48
  hdr_stats<<<gs, 256, 0, stream>>>(nr, rf, stats);
  hdr_reduce1<<<1, 256, 0, stream>>>(stats);
  hdr_mstat<<<gs, 256, 0, stream>>>(nr, stats);
  hdr_reduce2<<<1, 256, 0, stream>>>(stats);

  dim3 gm(Wn / (16 * TX), Hn / (16 * TY), Bn);       // 16 x 32 x 16
  hdr_mask<<<gm, 256, 0, stream>>>(nr, rf, stats, mA);

  int mblocks = Bn * PLANE / 256;                    // 65536
  hdr_morph_h<0><<<mblocks, 256, 0, stream>>>(mA, mB);
  hdr_morph_v<0><<<mblocks, 256, 0, stream>>>(mB, mA);
  hdr_morph_h<1><<<mblocks, 256, 0, stream>>>(mA, mB);
  hdr_final<<<mblocks, 256, 0, stream>>>(mB, out);
}
